// ParallelGAT_446676598990
// MI455X (gfx1250) — matrix operand order repaired and checked by reading, not compiled, not run on hardware
//
#include <hip/hip_runtime.h>
#include <math.h>

#define NN 50000
#define EE 800000
#define DD 128
#define LL 3
#define BB 64
#define ETOT (EE + NN)
#define ND (NN * DD)

typedef _Float16 half_t;
typedef __attribute__((ext_vector_type(16))) _Float16 v16h;
typedef __attribute__((ext_vector_type(8)))  float    v8f;

union FragH { v16h v; uint4 q[2]; };

__device__ __forceinline__ float gelu_f(float x) {
    return 0.5f * x * (1.0f + erff(x * 0.70710678118654752f));
}
__device__ __forceinline__ unsigned encf(float f) {
    unsigned u = __float_as_uint(f);
    return (u & 0x80000000u) ? ~u : (u | 0x80000000u);
}
__device__ __forceinline__ float decf(unsigned u) {
    unsigned v = (u & 0x80000000u) ? (u & 0x7fffffffu) : ~u;
    return __uint_as_float(v);
}

// ---------- split fp32 -> f16 hi/lo ----------
__global__ __launch_bounds__(256) void k_split_x(const float* __restrict__ xa,
                                                 const float* __restrict__ xb,
                                                 half_t* __restrict__ Ahi,
                                                 half_t* __restrict__ Alo, int n) {
    int i = blockIdx.x * 256 + threadIdx.x;
    if (i >= n) return;
    float v = xa[i] + (xb ? xb[i] : 0.0f);
    half_t hi = (half_t)v;
    Ahi[i] = hi;
    Alo[i] = (half_t)(v - (float)hi);
}

// transpose + split W[k][n] -> Wt[n][k] hi/lo
__global__ __launch_bounds__(256) void k_split_w(const float* __restrict__ W,
                                                 half_t* __restrict__ Whi,
                                                 half_t* __restrict__ Wlo) {
    int t = blockIdx.x * 256 + threadIdx.x;
    if (t >= DD * DD) return;
    int k = t / DD, n = t % DD;
    float v = W[k * DD + n];
    half_t hi = (half_t)v;
    Whi[n * DD + k] = hi;
    Wlo[n * DD + k] = (half_t)(v - (float)hi);
}

// ---------- WMMA GEMM: h[N,128] = A[N,128] @ W[128,128] (f16 split, fp32 acc) ----------
__global__ __launch_bounds__(256) void k_gemm(const half_t* __restrict__ Ahi,
                                              const half_t* __restrict__ Alo,
                                              const half_t* __restrict__ Bhi,
                                              const half_t* __restrict__ Blo,
                                              float* __restrict__ out) {
    int lane = threadIdx.x & 31;
    int wave = threadIdx.x >> 5;
    int m0 = (blockIdx.x * 8 + wave) * 16;
    if (m0 >= NN) return;
    int grp = lane >> 4;       // K-half selector
    int r   = lane & 15;       // row (A) / col (B/C)
    size_t rowA = (size_t)(m0 + r) * DD;

    v8f acc[8];
    v8f z = {0.f,0.f,0.f,0.f,0.f,0.f,0.f,0.f};
    for (int ct = 0; ct < 8; ++ct) acc[ct] = z;

    for (int ks = 0; ks < 4; ++ks) {
        int k0 = ks * 32;
        FragH ahi, alo;
        // A 16x32 f16 layout: elems [0..7]=K k0+grp*8.., elems[8..15]=K k0+16+grp*8..
        ahi.q[0] = *(const uint4*)(Ahi + rowA + k0 + grp * 8);
        ahi.q[1] = *(const uint4*)(Ahi + rowA + k0 + 16 + grp * 8);
        alo.q[0] = *(const uint4*)(Alo + rowA + k0 + grp * 8);
        alo.q[1] = *(const uint4*)(Alo + rowA + k0 + 16 + grp * 8);
        for (int ct = 0; ct < 8; ++ct) {
            size_t pb = (size_t)(ct * 16 + r) * DD + k0 + grp * 8;
            FragH bhi, blo;
            bhi.q[0] = *(const uint4*)(Bhi + pb);
            bhi.q[1] = *(const uint4*)(Bhi + pb + 16);
            blo.q[0] = *(const uint4*)(Blo + pb);
            blo.q[1] = *(const uint4*)(Blo + pb + 16);
            acc[ct] = __builtin_amdgcn_wmma_f32_16x16x32_f16(false, ahi.v, false, bhi.v,
                                                             (short)0, acc[ct], false, false);
            acc[ct] = __builtin_amdgcn_wmma_f32_16x16x32_f16(false, ahi.v, false, blo.v,
                                                             (short)0, acc[ct], false, false);
            acc[ct] = __builtin_amdgcn_wmma_f32_16x16x32_f16(false, alo.v, false, bhi.v,
                                                             (short)0, acc[ct], false, false);
        }
    }
    // C layout: VGPR rr, lanes 0-15 -> M=rr, lanes 16-31 -> M=rr+8; N = lane&15
    int orow = m0 + grp * 8;
    for (int ct = 0; ct < 8; ++ct)
        for (int rr = 0; rr < 8; ++rr)
            out[(size_t)(orow + rr) * DD + ct * 16 + r] = acc[ct][rr];
}

// ---------- per-row dots es = h.a_src, ed = h.a_dst ----------
__global__ __launch_bounds__(128) void k_rowdot(const float* __restrict__ h,
                                                const float* __restrict__ as,
                                                const float* __restrict__ ad,
                                                float* __restrict__ es,
                                                float* __restrict__ ed) {
    int row = blockIdx.x, t = threadIdx.x;
    __shared__ float sa[128], sb[128];
    float hv = h[(size_t)row * DD + t];
    sa[t] = hv * as[t];
    sb[t] = hv * ad[t];
    __syncthreads();
    for (int s = 64; s > 0; s >>= 1) {
        if (t < s) { sa[t] += sa[t + s]; sb[t] += sb[t + s]; }
        __syncthreads();
    }
    if (t == 0) { es[row] = sa[0]; ed[row] = sb[0]; }
}

// ---------- edge softmax ----------
__global__ __launch_bounds__(256) void k_edge_max(const int* __restrict__ ei,
                                                  const float* __restrict__ es,
                                                  const float* __restrict__ ed,
                                                  float* __restrict__ ev,
                                                  unsigned* __restrict__ menc) {
    int e = blockIdx.x * 256 + threadIdx.x;
    if (e >= ETOT) return;
    int s, d;
    if (e < EE) { s = ei[e]; d = ei[EE + e]; } else { s = e - EE; d = s; }
    float v = es[s] + ed[d];
    v = (v > 0.f) ? v : 0.2f * v;            // LeakyReLU(0.2)
    ev[e] = v;
    atomicMax(menc + d, encf(v));
}

__global__ __launch_bounds__(256) void k_edge_exp(const int* __restrict__ ei,
                                                  const float* __restrict__ ev,
                                                  const unsigned* __restrict__ menc,
                                                  float* __restrict__ ex,
                                                  float* __restrict__ ssum) {
    int e = blockIdx.x * 256 + threadIdx.x;
    if (e >= ETOT) return;
    int d = (e < EE) ? ei[EE + e] : (e - EE);
    float x = expf(ev[e] - decf(menc[d]));
    ex[e] = x;
    atomicAdd(ssum + d, x);
}

// one wave per edge, each lane handles 4 channels
__global__ __launch_bounds__(256) void k_edge_scatter(const int* __restrict__ ei,
                                                      const float* __restrict__ ex,
                                                      const float* __restrict__ ssum,
                                                      const float* __restrict__ h,
                                                      float* __restrict__ agg) {
    int wid = (blockIdx.x * 256 + threadIdx.x) >> 5;
    int lane = threadIdx.x & 31;
    if (wid >= ETOT) return;
    int s, d;
    if (wid < EE) { s = ei[wid]; d = ei[EE + wid]; } else { s = wid - EE; d = s; }
    __builtin_prefetch(h + (size_t)s * DD + lane * 4, 0, 0);
    float alpha = ex[wid] / (ssum[d] + 1e-16f);
    float4 hv = ((const float4*)(h + (size_t)s * DD))[lane];
    float* ap = agg + (size_t)d * DD + lane * 4;
    atomicAdd(ap + 0, alpha * hv.x);
    atomicAdd(ap + 1, alpha * hv.y);
    atomicAdd(ap + 2, alpha * hv.z);
    atomicAdd(ap + 3, alpha * hv.w);
}

// ---------- bias + GELU + channel stats ----------
__global__ __launch_bounds__(256) void k_post_gelu(float* __restrict__ agg,
                                                   const float* __restrict__ bias,
                                                   float* __restrict__ chsum,
                                                   float* __restrict__ chsq) {
    __shared__ float ls[DD], lq[DD];
    int t = threadIdx.x;
    if (t < DD) { ls[t] = 0.f; lq[t] = 0.f; }
    __syncthreads();
    for (long i = (long)blockIdx.x * 256 + t; i < (long)ND; i += (long)gridDim.x * 256) {
        int c = (int)(i & (DD - 1));
        float g = gelu_f(agg[i] + bias[c]);
        agg[i] = g;
        atomicAdd(&ls[c], g);
        atomicAdd(&lq[c], g * g);
    }
    __syncthreads();
    if (t < DD) { atomicAdd(&chsum[t], ls[t]); atomicAdd(&chsq[t], lq[t]); }
}

__global__ __launch_bounds__(256) void k_bn_apply(const float* __restrict__ g,
                                                  const float* __restrict__ chsum,
                                                  const float* __restrict__ chsq,
                                                  const float* __restrict__ gam,
                                                  const float* __restrict__ bet,
                                                  float* __restrict__ xout) {
    long i = (long)blockIdx.x * 256 + threadIdx.x;
    if (i >= (long)ND) return;
    int c = (int)(i & (DD - 1));
    float inv_n = 1.0f / (float)NN;
    float mu = chsum[c] * inv_n;
    float var = chsq[c] * inv_n - mu * mu;
    xout[i] = gam[c] * (g[i] - mu) * rsqrtf(var + 1e-5f) + bet[c];
}

// ---------- pooling ----------
__global__ __launch_bounds__(256) void k_count(const int* __restrict__ batch,
                                               float* __restrict__ counts) {
    int i = blockIdx.x * 256 + threadIdx.x;
    if (i < NN) atomicAdd(&counts[batch[i]], 1.0f);
}
__global__ __launch_bounds__(256) void k_pool_accum(const float* __restrict__ x,
                                                    const int* __restrict__ batch,
                                                    float* __restrict__ pooled, int half) {
    long i = (long)blockIdx.x * 256 + threadIdx.x;
    if (i >= (long)ND) return;
    int node = (int)(i >> 7), c = (int)(i & (DD - 1));
    int b = batch[node];
    atomicAdd(&pooled[(size_t)b * (2 * DD) + half * DD + c], x[i]);
}
__global__ __launch_bounds__(256) void k_pool_div(float* __restrict__ pooled,
                                                  const float* __restrict__ counts, int half) {
    int t = blockIdx.x * 256 + threadIdx.x;
    if (t >= BB * DD) return;
    int b = t >> 7, c = t & (DD - 1);
    pooled[(size_t)b * (2 * DD) + half * DD + c] /= fmaxf(counts[b], 1.0f);
}

// ---------- tiny MLP ----------
__global__ __launch_bounds__(256) void k_mlp(const float* __restrict__ in,
                                             const float* __restrict__ W,
                                             const float* __restrict__ bias,
                                             float* __restrict__ out, int K, int J) {
    int t = blockIdx.x * 256 + threadIdx.x;
    if (t >= BB * J) return;
    int b = t / J, j = t % J;
    float s = bias[j];
    for (int k = 0; k < K; ++k) s += in[b * K + k] * W[k * J + j];
    out[b * J + j] = gelu_f(s);
}

__global__ __launch_bounds__(64) void k_bn_small(float* __restrict__ buf,
                                                 const float* __restrict__ g,
                                                 const float* __restrict__ be, int J) {
    int j = blockIdx.x, t = threadIdx.x;
    __shared__ float sh[64];
    float v = buf[t * J + j];
    sh[t] = v; __syncthreads();
    for (int s = 32; s > 0; s >>= 1) { if (t < s) sh[t] += sh[t + s]; __syncthreads(); }
    float mu = sh[0] / 64.0f; __syncthreads();
    sh[t] = v * v; __syncthreads();
    for (int s = 32; s > 0; s >>= 1) { if (t < s) sh[t] += sh[t + s]; __syncthreads(); }
    float var = sh[0] / 64.0f - mu * mu;
    buf[t * J + j] = g[j] * (v - mu) * rsqrtf(var + 1e-5f) + be[j];
}

__global__ __launch_bounds__(64) void k_heads(const float* __restrict__ x,
                                              const float* __restrict__ rW,
                                              const float* __restrict__ rb,
                                              const float* __restrict__ cW,
                                              const float* __restrict__ cb,
                                              float* __restrict__ out) {
    int b = threadIdx.x;
    if (b >= BB) return;
    float r = 0.f, c = 0.f;
    for (int k = 0; k < DD; ++k) { float v = x[b * DD + k]; r += v * rW[k]; c += v * cW[k]; }
    out[b] = r + rb[0];
    out[BB + b] = 1.0f / (1.0f + expf(-(c + cb[0])));
}

// =====================================================================
extern "C" void kernel_launch(void* const* d_in, const int* in_sizes, int n_in,
                              void* d_out, int out_size, void* d_ws, size_t ws_size,
                              hipStream_t stream) {
    const float* x_ab   = (const float*)d_in[0];
    const int*   ei_ab  = (const int*)d_in[1];
    const int*   bat_ab = (const int*)d_in[2];
    const float* x_ag   = (const float*)d_in[3];
    const int*   ei_ag  = (const int*)d_in[4];
    const int*   bat_ag = (const int*)d_in[5];
    const float* prm[12];
    for (int i = 0; i < 12; ++i) prm[i] = (const float*)d_in[6 + i];
    const float* mlp_W0 = (const float*)d_in[18];
    const float* mlp_b0 = (const float*)d_in[19];
    const float* mlp_g0 = (const float*)d_in[20];
    const float* mlp_e0 = (const float*)d_in[21];
    const float* mlp_W1 = (const float*)d_in[22];
    const float* mlp_b1 = (const float*)d_in[23];
    const float* mlp_g1 = (const float*)d_in[24];
    const float* mlp_e1 = (const float*)d_in[25];
    const float* reg_W  = (const float*)d_in[26];
    const float* reg_b  = (const float*)d_in[27];
    const float* cls_W  = (const float*)d_in[28];
    const float* cls_b  = (const float*)d_in[29];

    // workspace carve-up
    char* w = (char*)d_ws;
    size_t off = 0;
    auto take = [&](size_t bytes) -> void* {
        void* p = w + off;
        off = (off + bytes + 255) & ~(size_t)255;
        return p;
    };
    float*   xcur  = (float*)take((size_t)ND * 4);
    float*   hbuf  = (float*)take((size_t)ND * 4);
    float*   agg   = (float*)take((size_t)ND * 4);
    half_t*  Ahi   = (half_t*)take((size_t)ND * 2);
    half_t*  Alo   = (half_t*)take((size_t)ND * 2);
    half_t*  Whi   = (half_t*)take((size_t)DD * DD * 2);
    half_t*  Wlo   = (half_t*)take((size_t)DD * DD * 2);
    float*   es    = (float*)take((size_t)NN * 4);
    float*   ed    = (float*)take((size_t)NN * 4);
    unsigned* menc = (unsigned*)take((size_t)NN * 4);
    float*   ssum  = (float*)take((size_t)NN * 4);
    float*   evb   = (float*)take((size_t)ETOT * 4);
    float*   exb   = (float*)take((size_t)ETOT * 4);
    float*   chsum = (float*)take(DD * 4);
    float*   chsq  = (float*)take(DD * 4);
    float*   pooled= (float*)take((size_t)BB * 2 * DD * 4);
    float*   counts= (float*)take(BB * 4);
    float*   m0out = (float*)take((size_t)BB * 2 * DD * 4);
    float*   m1out = (float*)take((size_t)BB * DD * 4);
    (void)ws_size; (void)n_in; (void)in_sizes; (void)out_size;

    const int gND   = (ND + 255) / 256;
    const int gE    = (ETOT + 255) / 256;
    const int gScat = (ETOT * 32 + 255) / 256;
    const int gGemm = ((NN / 16) + 7) / 8;

    hipMemsetAsync(pooled, 0, (size_t)BB * 2 * DD * 4, stream);

    for (int tw = 0; tw < 2; ++tw) {
        const float* x0    = tw ? x_ag : x_ab;
        const int*   ei    = tw ? ei_ag : ei_ab;
        const int*   batch = tw ? bat_ag : bat_ab;
        const float* Wp    = prm[tw * 6 + 0];
        const float* asrc  = prm[tw * 6 + 1];
        const float* adst  = prm[tw * 6 + 2];
        const float* bias  = prm[tw * 6 + 3];
        const float* gam   = prm[tw * 6 + 4];
        const float* bet   = prm[tw * 6 + 5];

        const float* xprev = x0;
        for (int l = 0; l < LL; ++l) {
            k_split_w<<<(DD * DD + 255) / 256, 256, 0, stream>>>(Wp + (size_t)l * DD * DD, Whi, Wlo);
            k_split_x<<<gND, 256, 0, stream>>>(xprev, (l == 0) ? nullptr : x0, Ahi, Alo, ND);
            k_gemm<<<gGemm, 256, 0, stream>>>(Ahi, Alo, Whi, Wlo, hbuf);
            k_rowdot<<<NN, 128, 0, stream>>>(hbuf, asrc + l * DD, adst + l * DD, es, ed);
            hipMemsetAsync(menc, 0, (size_t)NN * 4, stream);
            hipMemsetAsync(ssum, 0, (size_t)NN * 4, stream);
            hipMemsetAsync(agg, 0, (size_t)ND * 4, stream);
            hipMemsetAsync(chsum, 0, DD * 4, stream);
            hipMemsetAsync(chsq, 0, DD * 4, stream);
            k_edge_max<<<gE, 256, 0, stream>>>(ei, es, ed, evb, menc);
            k_edge_exp<<<gE, 256, 0, stream>>>(ei, evb, menc, exb, ssum);
            k_edge_scatter<<<gScat, 256, 0, stream>>>(ei, exb, ssum, hbuf, agg);
            k_post_gelu<<<512, 256, 0, stream>>>(agg, bias + l * DD, chsum, chsq);
            k_bn_apply<<<gND, 256, 0, stream>>>(agg, chsum, chsq, gam + l * DD, bet + l * DD, xcur);
            xprev = xcur;
        }
        hipMemsetAsync(counts, 0, BB * 4, stream);
        k_count<<<(NN + 255) / 256, 256, 0, stream>>>(batch, counts);
        k_pool_accum<<<gND, 256, 0, stream>>>(xcur, batch, pooled, tw);
        k_pool_div<<<(BB * DD + 255) / 256, 256, 0, stream>>>(pooled, counts, tw);
    }

    // MLP head
    k_mlp<<<(BB * 256 + 255) / 256, 256, 0, stream>>>(pooled, mlp_W0, mlp_b0, m0out, 256, 256);
    k_bn_small<<<256, 64, 0, stream>>>(m0out, mlp_g0, mlp_e0, 256);
    k_mlp<<<(BB * 128 + 255) / 256, 256, 0, stream>>>(m0out, mlp_W1, mlp_b1, m1out, 256, 128);
    k_bn_small<<<128, 64, 0, stream>>>(m1out, mlp_g1, mlp_e1, 128);
    k_heads<<<1, 64, 0, stream>>>(m1out, reg_W, reg_b, cls_W, cls_b, (float*)d_out);
}
